// KModule_65824668778526
// MI455X (gfx1250) — matrix operand order repaired and checked by reading, not compiled, not run on hardware
//
#include <hip/hip_runtime.h>
#include <hip/hip_bf16.h>
#include <math.h>

// Problem constants (from reference)
#define BB 4
#define NN 4096
#define MM 4096
#define DD 256
#define RR 64
#define KK 16

typedef _Float16 v16h __attribute__((ext_vector_type(16)));
typedef float    v8f  __attribute__((ext_vector_type(8)));
typedef unsigned int u32x4 __attribute__((ext_vector_type(4)));
typedef int          i32x4 __attribute__((ext_vector_type(4)));
typedef int          i32x8 __attribute__((ext_vector_type(8)));

#ifndef __has_builtin
#define __has_builtin(x) 0
#endif
#if __has_builtin(__builtin_amdgcn_tensor_load_to_lds) && __has_builtin(__builtin_amdgcn_s_wait_tensorcnt)
#define HAVE_TDM 1
#else
#define HAVE_TDM 0
#endif

#if HAVE_TDM
// Tensor Data Mover: async DMA of one 16-row x 64-halfword tile (2KB) into LDS.
// D# per CDNA5 ISA 8.3/8.4: group0 = {count/type, lds_addr, global_addr lo/hi},
// group1 = {data_size=2B, tensor_dim0=64, tensor_dim1=16, tile 64x16, stride 64}.
// Tracked with TENSORcnt (in-order per wave). Descriptor components are forced
// wave-uniform via readfirstlane (they are loaded into SGPR groups).
__device__ __forceinline__ void tdm_load_tile16x64(const _Float16* gsrc, unsigned lds_byte_addr) {
  const unsigned long long ga = (unsigned long long)gsrc;
  u32x4 g0;
  g0[0] = 1u;  // descriptor count = 1, user mode
  g0[1] = (unsigned)__builtin_amdgcn_readfirstlane((int)lds_byte_addr);
  g0[2] = (unsigned)__builtin_amdgcn_readfirstlane((int)(unsigned)(ga & 0xFFFFFFFFull));
  g0[3] = ((unsigned)__builtin_amdgcn_readfirstlane((int)(unsigned)(ga >> 32)) & 0x1FFFFFFu)
          | 0x80000000u;  // type = 2 ("image") in bits 127:126
  i32x8 g1 = {0, 0, 0, 0, 0, 0, 0, 0};
  g1[0] = 1 << 16;    // workgroup_mask=0 (not in cluster), data_size=1 -> 2 bytes
  g1[1] = RR << 16;   // tensor_dim0 = 64 elements      (bits 79:48)
  g1[2] = 16 << 16;   // tensor_dim1 = 16 rows          (bits 111:80)
  g1[3] = RR << 16;   // tile_dim0   = 64               (bits 127:112)
  g1[4] = 16;         // tile_dim1   = 16               (bits 143:128), tile_dim2 = 0
  g1[5] = RR;         // tensor_dim0_stride = 64        (bits 191:160)
  const i32x4 gz = {0, 0, 0, 0};
#if __clang_major__ >= 23
  const i32x8 gz8 = {0, 0, 0, 0, 0, 0, 0, 0};
  __builtin_amdgcn_tensor_load_to_lds(g0, g1, gz, gz, gz8, 0);
#else
  __builtin_amdgcn_tensor_load_to_lds(g0, g1, gz, gz, 0);
#endif
}
#endif

// ---------------------------------------------------------------------------
// signed softmax over last axis (length n) per batch row b = blockIdx.x
// ---------------------------------------------------------------------------
__global__ void ssm_kernel(const float* __restrict__ x, const float* __restrict__ dx,
                           float* __restrict__ out, int n, int bcast) {
  __shared__ float red[256];
  const int b = blockIdx.x, t = threadIdx.x;
  const float* xr = bcast ? x : x + (size_t)b * n;
  const float* dr = dx ? dx + (size_t)b * n : nullptr;
  float mx = -1e30f;
  for (int i = t; i < n; i += 256) {
    float v = xr[i]; if (dr) v += dr[i];
    mx = fmaxf(mx, fabsf(v));
  }
  red[t] = mx; __syncthreads();
  for (int s = 128; s > 0; s >>= 1) { if (t < s) red[t] = fmaxf(red[t], red[t + s]); __syncthreads(); }
  mx = red[0]; __syncthreads();
  float sm = 0.f;
  for (int i = t; i < n; i += 256) {
    float v = xr[i]; if (dr) v += dr[i];
    sm += __expf(fabsf(v) - mx);
  }
  red[t] = sm; __syncthreads();
  for (int s = 128; s > 0; s >>= 1) { if (t < s) red[t] += red[t + s]; __syncthreads(); }
  const float inv = 1.f / red[0];
  for (int i = t; i < n; i += 256) {
    float v = xr[i]; if (dr) v += dr[i];
    out[(size_t)b * n + i] = copysignf(__expf(fabsf(v) - mx) * inv, v);
  }
}

// ---------------------------------------------------------------------------
// layernorm over D=256 per row; one wave32 per row, 8 elements per lane.
// ---------------------------------------------------------------------------
__global__ void layernorm_kernel(const float* __restrict__ x, const float* __restrict__ dx,
                                 const float* __restrict__ g, const float* __restrict__ bta,
                                 float* __restrict__ out, int rows, int bmod) {
  const int row = blockIdx.x * 8 + (threadIdx.x >> 5);
  const int lane = threadIdx.x & 31;
  if (row >= rows) return;
  const size_t xr = (size_t)(bmod ? (row % bmod) : row) * DD;
  const size_t orow = (size_t)row * DD;
  float v[8]; float s = 0.f, s2 = 0.f;
#pragma unroll
  for (int i = 0; i < 8; ++i) {
    float t = x[xr + lane + 32 * i];
    if (dx) t += dx[orow + lane + 32 * i];
    v[i] = t; s += t; s2 += t * t;
  }
#pragma unroll
  for (int m = 16; m >= 1; m >>= 1) { s += __shfl_xor(s, m, 32); s2 += __shfl_xor(s2, m, 32); }
  const float mean = s * (1.f / DD);
  const float var = s2 * (1.f / DD) - mean * mean;
  const float r = rsqrtf(var + 1e-5f);
#pragma unroll
  for (int i = 0; i < 8; ++i) {
    const int d = lane + 32 * i;
    out[orow + d] = (v[i] - mean) * r * g[d] + bta[d];
  }
}

// ---------------------------------------------------------------------------
// row L2 norms over D=256; one wave32 per row.
// ---------------------------------------------------------------------------
__global__ void norms_kernel(const float* __restrict__ x, float* __restrict__ nrm, int rows) {
  const int row = blockIdx.x * 8 + (threadIdx.x >> 5);
  const int lane = threadIdx.x & 31;
  if (row >= rows) return;
  float s = 0.f;
#pragma unroll
  for (int i = 0; i < 8; ++i) { float v = x[(size_t)row * DD + lane + 32 * i]; s += v * v; }
#pragma unroll
  for (int m = 16; m >= 1; m >>= 1) s += __shfl_xor(s, m, 32);
  if (lane == 0) nrm[row] = sqrtf(s);
}

// ---------------------------------------------------------------------------
// Projection GEMM with WMMA f16: P[rows,64] = f16( X[rows,256] @ W[256,64] * scale )
// Block = 128 threads (4 waves); wave w handles column chunk [16w, 16w+16).
// ---------------------------------------------------------------------------
__global__ void proj_kernel(const float* __restrict__ X, const float* __restrict__ W,
                            _Float16* __restrict__ P, float scale) {
  const int wvid = threadIdx.x >> 5, lane = threadIdx.x & 31;
  const int rowbase = blockIdx.x * 16;
  const int colbase = wvid * 16;
  const int half = lane >> 4;
  const int arow = rowbase + (lane & 15);
  const int bcol = colbase + (lane & 15);
  v8f c = {0.f, 0.f, 0.f, 0.f, 0.f, 0.f, 0.f, 0.f};
#pragma unroll
  for (int kb = 0; kb < 8; ++kb) {
    v16h a, bm;
#pragma unroll
    for (int j = 0; j < 8; ++j) {
      const int ka = kb * 32 + (j >> 2) * 16 + half * 8 + (j & 3) * 2;  // A-frag K pair
      a[2 * j]     = (_Float16)X[(size_t)arow * DD + ka];
      a[2 * j + 1] = (_Float16)X[(size_t)arow * DD + ka + 1];
      const int kbx = kb * 32 + (j >> 2) * 16 + half * 8 + (j & 3) * 2;  // B-frag K pair
      bm[2 * j]     = (_Float16)W[(size_t)kbx * RR + bcol];
      bm[2 * j + 1] = (_Float16)W[(size_t)(kbx + 1) * RR + bcol];
    }
    c = __builtin_amdgcn_wmma_f32_16x16x32_f16(false, a, false, bm, (short)0, c, false, false);
  }
#pragma unroll
  for (int v = 0; v < 8; ++v) {
    const int r = rowbase + half * 8 + v;
    P[(size_t)r * RR + bcol] = (_Float16)(c[v] * scale);
  }
}

// ---------------------------------------------------------------------------
// Fused score GEMM + top-K(|.|) + signed softmax (+ softplus(src_state)).
// scores[b, n_dst, m_src] = dp[n_dst] . sp[m_src]   (1/sqrt(R) folded into dp)
// A = sp tile (src rows): TDM double-buffered into per-wave LDS buffers
//     (fallback: manual b128 copy + block barriers when TDM builtin absent).
// B = dp rows (dst cols), VGPR-resident for the whole kernel.
// C layout puts all 8 accum registers of a lane on ONE dst row -> per-lane
// register top-16 with min-threshold filter; lanes l / l+16 merged via LDS.
// Grid: (NN/64, BB), block = 128 (4 waves, wave w -> dst tile 4*bx + w).
// ---------------------------------------------------------------------------
__global__ void score_topk_kernel(const _Float16* __restrict__ sp, const _Float16* __restrict__ dp,
                                  const float* __restrict__ src_state,
                                  float* __restrict__ wts, int* __restrict__ idxo,
                                  float* __restrict__ dstate) {
  __shared__ __align__(16) _Float16 stage[4][2][16 * RR];  // 4 waves x double buffer x 2KB
  __shared__ float mv[4][512];
  __shared__ int   mi[4][512];
  const int wvid = threadIdx.x >> 5, lane = threadIdx.x & 31;
  const int bb = blockIdx.y;
  const int n0 = (blockIdx.x * 4 + wvid) * 16;
  const int half = lane >> 4;
  const int mhalf = half * 8;

  // B fragments: dp rows n0..n0+15, K = 0..63 (two 32-deep fragments), resident.
  v16h bf0, bf1;
  {
    const _Float16* dpr = dp + (size_t)(bb * NN + n0 + (lane & 15)) * RR;
#pragma unroll
    for (int j = 0; j < 8; ++j) {
      const int k0 = (j >> 2) * 16 + half * 8 + (j & 3) * 2;
      bf0[2 * j] = dpr[k0];      bf0[2 * j + 1] = dpr[k0 + 1];
      bf1[2 * j] = dpr[32 + k0]; bf1[2 * j + 1] = dpr[32 + k0 + 1];
    }
  }

  float tv[KK]; int ti[KK]; float thr = -1.f;
#pragma unroll
  for (int k = 0; k < KK; ++k) { tv[k] = 0.f; ti[k] = 0; }

  const _Float16* spb = sp + (size_t)bb * MM * RR;
  const int T = MM / 16;

#if HAVE_TDM
  const unsigned lds0 = (unsigned)(unsigned long long)(const void*)&stage[wvid][0][0];
  const unsigned lds1 = (unsigned)(unsigned long long)(const void*)&stage[wvid][1][0];
  tdm_load_tile16x64(spb, lds0);  // preload tile 0
#endif

  for (int mt = 0; mt < T; ++mt) {
    const int m0 = mt * 16;
    const int cur = mt & 1;
    const _Float16* sr0;
#if HAVE_TDM
    // Issue DMA for the next tile into the other buffer, then wait until only
    // that one is outstanding (TDM is in-order per wave => current tile ready).
    tdm_load_tile16x64(spb + (size_t)((mt + 1) & (T - 1)) * 16 * RR, cur ? lds0 : lds1);
    __builtin_amdgcn_s_wait_tensorcnt(1);
    sr0 = &stage[wvid][cur][0];
#else
    (void)cur;
    const uint4* srcp = (const uint4*)(spb + (size_t)m0 * RR);
    uint4* dstp = (uint4*)&stage[wvid][0][0];
#pragma unroll
    for (int i = 0; i < 4; ++i) dstp[lane + 32 * i] = srcp[lane + 32 * i];
    if (mt + 1 < T) __builtin_prefetch(spb + (size_t)(m0 + 16) * RR, 0, 1);
    __syncthreads();
    sr0 = &stage[wvid][0][0];
#endif
    // A fragments from LDS.
    v16h a0, a1;
    {
      const _Float16* sr = sr0 + (lane & 15) * RR;
#pragma unroll
      for (int j = 0; j < 8; ++j) {
        const int k0 = (j >> 2) * 16 + half * 8 + (j & 3) * 2;
        a0[2 * j] = sr[k0];      a0[2 * j + 1] = sr[k0 + 1];
        a1[2 * j] = sr[32 + k0]; a1[2 * j + 1] = sr[32 + k0 + 1];
      }
    }
#if !HAVE_TDM
    __syncthreads();
#endif
    v8f c = {0.f, 0.f, 0.f, 0.f, 0.f, 0.f, 0.f, 0.f};
    c = __builtin_amdgcn_wmma_f32_16x16x32_f16(false, a0, false, bf0, (short)0, c, false, false);
    c = __builtin_amdgcn_wmma_f32_16x16x32_f16(false, a1, false, bf1, (short)0, c, false, false);
    // Top-K maintenance: lane's 8 scores are src rows m0+mhalf+v of its dst row.
#pragma unroll
    for (int v = 0; v < 8; ++v) {
      const float s = c[v];
      if (fabsf(s) > thr) {
        int p = 0; float pm = fabsf(tv[0]);
#pragma unroll
        for (int k = 1; k < KK; ++k) { const float ak = fabsf(tv[k]); if (ak < pm) { pm = ak; p = k; } }
        const int midx = m0 + mhalf + v;
#pragma unroll
        for (int k = 0; k < KK; ++k) { const bool r = (k == p); tv[k] = r ? s : tv[k]; ti[k] = r ? midx : ti[k]; }
        thr = fabsf(tv[0]);
#pragma unroll
        for (int k = 1; k < KK; ++k) thr = fminf(thr, fabsf(tv[k]));
      }
    }
  }

  // Merge lane pairs (d, d+16) -> final top-16 per dst row, softmax, outputs.
#pragma unroll
  for (int k = 0; k < KK; ++k) { mv[wvid][lane * 16 + k] = tv[k]; mi[wvid][lane * 16 + k] = ti[k]; }
  __syncthreads();
  if (lane < 16) {
    const int n = n0 + lane;
    unsigned taken = 0u;
    float wv_[KK]; int wi_[KK];
#pragma unroll
    for (int k = 0; k < KK; ++k) {
      float best = -1.f; int bp = 0;
      for (int j = 0; j < 32; ++j) {
        if (!((taken >> j) & 1u)) {
          const float cv = mv[wvid][(lane + ((j >> 4) << 4)) * 16 + (j & 15)];
          const float av = fabsf(cv);
          if (av > best) { best = av; bp = j; }
        }
      }
      taken |= 1u << bp;
      wv_[k] = mv[wvid][(lane + ((bp >> 4) << 4)) * 16 + (bp & 15)];
      wi_[k] = mi[wvid][(lane + ((bp >> 4) << 4)) * 16 + (bp & 15)];
    }
    const float mx = fabsf(wv_[0]);
    float e[KK], S = 0.f;
#pragma unroll
    for (int k = 0; k < KK; ++k) { e[k] = __expf(fabsf(wv_[k]) - mx); S += e[k]; }
    const float inv = 1.f / S;
    float dsum = 0.f;
    const size_t ob = (size_t)(bb * NN + n) * KK;
#pragma unroll
    for (int k = 0; k < KK; ++k) {
      float w = copysignf(e[k] * inv, wv_[k]);
      if (src_state) {
        const float xs = src_state[(size_t)bb * MM + wi_[k]];
        w *= (xs > 20.f) ? xs : log1pf(__expf(xs));  // softplus
      }
      wts[ob + k] = w; idxo[ob + k] = wi_[k]; dsum += w;
    }
    dstate[(size_t)bb * NN + n] = dsum;
  }
}

// ---------------------------------------------------------------------------
// dval[b,n,:] = sum_k wts[k] * src_val[b, idx[k], :] / (||src_val[b,idx[k]]|| + 1e-6)
// ---------------------------------------------------------------------------
__global__ void gather_kernel(const float* __restrict__ srcval, const float* __restrict__ nrm,
                              const float* __restrict__ wts, const int* __restrict__ idxs,
                              float* __restrict__ dval) {
  __shared__ float wl[KK];
  __shared__ int il[KK];
  const int n = blockIdx.x, bb = blockIdx.y, d = threadIdx.x;
  if (d < KK) {
    const size_t ob = (size_t)(bb * NN + n) * KK + d;
    const int id = idxs[ob];
    il[d] = id;
    wl[d] = wts[ob] / (nrm[(size_t)bb * MM + id] + 1e-6f);
  }
  __syncthreads();
  float acc = 0.f;
#pragma unroll
  for (int k = 0; k < KK; ++k) acc += wl[k] * srcval[((size_t)bb * MM + il[k]) * DD + d];
  dval[((size_t)bb * NN + n) * DD + d] = acc;
}

// ---------------------------------------------------------------------------
extern "C" void kernel_launch(void* const* d_in, const int* in_sizes, int n_in,
                              void* d_out, int out_size, void* d_ws, size_t ws_size,
                              hipStream_t stream) {
  (void)in_sizes; (void)n_in; (void)out_size; (void)ws_size;
  const float* b_state    = (const float*)d_in[0];
  const float* b_val      = (const float*)d_in[1];
  const float* init_state = (const float*)d_in[2];
  const float* init_val   = (const float*)d_in[3];
  const float* U_bk = (const float*)d_in[4];
  const float* V_bk = (const float*)d_in[5];
  const float* U_kb = (const float*)d_in[6];
  const float* V_kb = (const float*)d_in[7];
  const float* U_p  = (const float*)d_in[8];
  const float* V_p  = (const float*)d_in[9];
  const float* kn_g = (const float*)d_in[10];
  const float* kn_b = (const float*)d_in[11];
  const float* bn_g = (const float*)d_in[12];
  const float* bn_b = (const float*)d_in[13];
  const float* pn_g = (const float*)d_in[14];
  const float* pn_b = (const float*)d_in[15];
  float* out = (float*)d_out;

  // d_out slice offsets (floats): r_state, r_val, p_state, p_val, bds, bdv
  const size_t O_RS = 0, O_RV = 16384, O_PS = 4210688, O_PV = 4227072, O_BDS = 8421376, O_BDV = 8437760;

  // Workspace carve-out
  char* w = (char*)d_ws;
  auto alloc = [&](size_t bytes) { char* p = w; w += (bytes + 255) & ~(size_t)255; return p; };
  float*     kstate = (float*)alloc((size_t)BB * NN * 4);
  float*     kval   = (float*)alloc((size_t)BB * NN * DD * 4);
  float*     nk     = (float*)alloc((size_t)BB * NN * DD * 4);  // reused as nk2
  float*     nv     = (float*)alloc((size_t)BB * NN * DD * 4);
  float*     nb     = (float*)alloc((size_t)BB * NN * DD * 4);
  float*     dval   = (float*)alloc((size_t)BB * NN * DD * 4);
  _Float16*  sp     = (_Float16*)alloc((size_t)BB * MM * RR * 2);
  _Float16*  dp     = (_Float16*)alloc((size_t)BB * NN * RR * 2);
  float*     wts    = (float*)alloc((size_t)BB * NN * KK * 4);
  int*       idx    = (int*)alloc((size_t)BB * NN * KK * 4);
  float*     nrm    = (float*)alloc((size_t)BB * MM * 4);
  float*     dst    = (float*)alloc((size_t)BB * NN * 4);

  const dim3 lnG(2048), lnB(256);         // wave-per-row kernels over 16384 rows
  const dim3 pjG(1024), pjB(128);         // projection GEMM
  const dim3 scG(64, 4), scB(128);        // fused score+topk
  const dim3 gaG(4096, 4), gaB(256);      // gather

  // ---- initialize_state ----
  ssm_kernel<<<BB, 256, 0, stream>>>(init_state, nullptr, kstate, NN, 1);
  layernorm_kernel<<<lnG, lnB, 0, stream>>>(init_val, nullptr, kn_g, kn_b, kval, BB * NN, NN);

  // ---- route_from_b: B -> K ----
  layernorm_kernel<<<lnG, lnB, 0, stream>>>(kval, nullptr, kn_g, kn_b, nk, BB * NN, 0);
  proj_kernel<<<pjG, pjB, 0, stream>>>(b_val, U_bk, sp, 1.0f);
  proj_kernel<<<pjG, pjB, 0, stream>>>(nk, V_bk, dp, 0.125f);  // 1/sqrt(R)
  score_topk_kernel<<<scG, scB, 0, stream>>>(sp, dp, b_state, wts, idx, dst);
  norms_kernel<<<lnG, lnB, 0, stream>>>(b_val, nrm, BB * MM);
  gather_kernel<<<gaG, gaB, 0, stream>>>(b_val, nrm, wts, idx, dval);
  ssm_kernel<<<BB, 256, 0, stream>>>(kstate, dst, out + O_RS, NN, 0);
  layernorm_kernel<<<lnG, lnB, 0, stream>>>(kval, dval, kn_g, kn_b, out + O_RV, BB * NN, 0);

  // ---- propagate ----
  layernorm_kernel<<<lnG, lnB, 0, stream>>>(out + O_RV, nullptr, pn_g, pn_b, nv, BB * NN, 0);
  proj_kernel<<<pjG, pjB, 0, stream>>>(nv, U_p, sp, 1.0f);
  proj_kernel<<<pjG, pjB, 0, stream>>>(nv, V_p, dp, 0.125f);
  score_topk_kernel<<<scG, scB, 0, stream>>>(sp, dp, nullptr, wts, idx, dst);
  norms_kernel<<<lnG, lnB, 0, stream>>>(nv, nrm, BB * NN);
  gather_kernel<<<gaG, gaB, 0, stream>>>(nv, nrm, wts, idx, dval);
  ssm_kernel<<<BB, 256, 0, stream>>>(out + O_RS, dst, out + O_PS, NN, 0);
  layernorm_kernel<<<lnG, lnB, 0, stream>>>(out + O_RV, dval, kn_g, kn_b, out + O_PV, BB * NN, 0);

  // ---- transition_to_b: K -> B (delta only) ----
  layernorm_kernel<<<lnG, lnB, 0, stream>>>(out + O_PV, nullptr, kn_g, kn_b, nk, BB * NN, 0);  // nk2
  layernorm_kernel<<<lnG, lnB, 0, stream>>>(b_val, nullptr, bn_g, bn_b, nb, BB * NN, 0);
  proj_kernel<<<pjG, pjB, 0, stream>>>(nk, U_kb, sp, 1.0f);
  proj_kernel<<<pjG, pjB, 0, stream>>>(nb, V_kb, dp, 0.125f);
  score_topk_kernel<<<scG, scB, 0, stream>>>(sp, dp, out + O_PS, wts, idx, out + O_BDS);
  norms_kernel<<<lnG, lnB, 0, stream>>>(nk, nrm, BB * NN);
  gather_kernel<<<gaG, gaB, 0, stream>>>(nk, nrm, wts, idx, out + O_BDV);
}
